// TransformerBlock_47725676593725
// MI455X (gfx1250) — matrix operand order repaired and checked by reading, not compiled, not run on hardware
//
#include <hip/hip_runtime.h>
#include <hip/hip_bf16.h>

// ---------------------------------------------------------------------------
// TransformerBlock for MI455X (gfx1250): all matmuls via v_wmma_f32_16x16x32_f16
// ---------------------------------------------------------------------------

#define B_ 4
#define S_ 2048
#define E_ 1024
#define H_ 16
#define DH_ 64
#define M_ (B_ * S_)   // 8192 rows
#define EPS_ 1e-5f

typedef _Float16 v16h __attribute__((ext_vector_type(16)));
typedef _Float16 h8   __attribute__((ext_vector_type(8)));
typedef float    v8f  __attribute__((ext_vector_type(8)));

union H16 { v16h v; h8 h[2]; };

static __device__ inline v8f wmma_f16(v16h a, v16h b, v8f c) {
    // D = A(16x32 f16) * B(32x16 f16) + C(16x16 f32)
    return __builtin_amdgcn_wmma_f32_16x16x32_f16(
        /*neg_a=*/false, a, /*neg_b=*/false, b,
        /*c_mod=*/(short)0, c, /*reuse_a=*/false, /*reuse_b=*/false);
}

// ------------------------------- f32 -> f16 --------------------------------
__global__ __launch_bounds__(256) void cvt_f32_f16(const float* __restrict__ in,
                                                   _Float16* __restrict__ out,
                                                   long n) {
    long i = (long)blockIdx.x * 256 + threadIdx.x;
    if (i < n) out[i] = (_Float16)in[i];
}

// ----------------------------- WMMA GEMM -----------------------------------
// C(M,N) = A(M,K) * W(N,K)^T + bias
// Wave tile 64x64 (4 A-frags x 4 B-frags -> 16 WMMAs / k-step), block = 8
// waves covering 128x256.  OUT_F16: store f16, else f32.  RELU: ReLU epilogue.
// TRANS_V: scatter-store as (B,H,Dh,S) (for V so attention B-frags are linear).
template <int OUT_F16, int RELU, int TRANS_V>
__global__ __launch_bounds__(256) void gemm_wmma(const _Float16* __restrict__ A,
                                                 const _Float16* __restrict__ W,
                                                 const float* __restrict__ bias,
                                                 void* __restrict__ outp,
                                                 int M, int N, int K) {
    const int lane = threadIdx.x & 31;
    const int wave = threadIdx.x >> 5;
    const int wm = wave >> 2;                 // 0..1
    const int wn = wave & 3;                  // 0..3
    const int m0 = blockIdx.y * 128 + wm * 64;
    const int n0 = blockIdx.x * 256 + wn * 64;
    const int l16 = lane & 15;
    const int hsel = (lane < 16) ? 0 : 1;     // lane half select
    const int kA = hsel * 16;                 // A-frag k base (contiguous 16)
    const int kB = hsel * 16;                 // B-frag k base (contiguous 16)

    v8f acc[4][4];
#pragma unroll
    for (int i = 0; i < 4; i++)
#pragma unroll
        for (int j = 0; j < 4; j++) acc[i][j] = (v8f)0.0f;

    long rowA[4];
#pragma unroll
    for (int i = 0; i < 4; i++) rowA[i] = (long)(m0 + i * 16 + l16) * K;

    for (int kk = 0; kk < K; kk += 32) {
        H16 a[4];
#pragma unroll
        for (int i = 0; i < 4; i++) {
            const h8* pa = (const h8*)(A + rowA[i] + kk + kA);
            a[i].h[0] = pa[0]; a[i].h[1] = pa[1];   // k = kA+{0..15}
        }
#pragma unroll
        for (int j = 0; j < 4; j++) {
            const int n = n0 + j * 16 + l16;
            v16h bf = *(const v16h*)(W + (long)n * K + kk + kB);
#pragma unroll
            for (int i = 0; i < 4; i++)
                acc[i][j] = wmma_f16(a[i].v, bf, acc[i][j]);
        }
    }

    // Epilogue.  C/D layout: vgpr r, lane L -> m_local = r + 8*(L>=16), n = L%16
#pragma unroll
    for (int i = 0; i < 4; i++) {
#pragma unroll
        for (int j = 0; j < 4; j++) {
            const int n = n0 + j * 16 + l16;
            const float bv = bias ? bias[n] : 0.0f;
#pragma unroll
            for (int r = 0; r < 8; r++) {
                const int m = m0 + i * 16 + r + hsel * 8;
                float v = acc[i][j][r] + bv;
                if (RELU) v = v > 0.0f ? v : 0.0f;
                if (OUT_F16) {
                    _Float16* o = (_Float16*)outp;
                    if (TRANS_V) {
                        const int s = m & (S_ - 1), b = m >> 11;   // S_=2048
                        const int hh = n >> 6, d = n & 63;          // Dh=64
                        o[(((long)b * H_ + hh) * DH_ + d) * S_ + s] = (_Float16)v;
                    } else {
                        o[(long)m * N + n] = (_Float16)v;
                    }
                } else {
                    ((float*)outp)[(long)m * N + n] = v;
                }
            }
        }
    }
}

// --------------------------- Attention -------------------------------------
// One wave per (b, h, 32-row query tile); 32-key chunks.
// Softmax without max-shift (scores are O(1) for this model), row sums kept as
// per-thread partials and reduced ONCE after the K loop -> no per-chunk
// cross-lane reductions.  Both 1/sqrt(Dh) AND log2(e) folded into the Q
// fragments so the hot loop uses raw v_exp_f32 (exp2) with no scaling muls.
// Per chunk: 8 score WMMAs + 8 PV WMMAs for 8 K/V v16h loads.
// Launched as single-wave workgroups: barrier ops are S_NOP per the ISA, so
// the __syncthreads() around the P LDS transpose is free.
__global__ __launch_bounds__(32) void attn_wmma(const _Float16* __restrict__ Q,
                                                const _Float16* __restrict__ Km,
                                                const _Float16* __restrict__ Vt,
                                                _Float16* __restrict__ O) {
    __shared__ _Float16 lds_p[2][16 * 32];    // P tiles in A-layout staging
    const int lane = threadIdx.x & 31;
    const int l16 = lane & 15;
    const int kAoff = (lane < 16) ? 0 : 16;
    const int kBoff = (lane < 16) ? 0 : 16;
    const int mrow = (lane < 16) ? 0 : 8;

    const int nQT = S_ / 32;                  // 64 query tiles per (b,h)
    const int task = blockIdx.x;              // one wave per task
    const int bh = task / nQT;
    const int qt = task % nQT;
    const int b = bh / H_, h = bh % H_;
    const int q0 = qt * 32;

    // Q fragments for 2 query sub-tiles x 2 d-halves, scaled by log2(e)/8 so
    // scores come out in base-2 -> exp2 is a single v_exp_f32.
    H16 aq[2][2];
    {
        const v16h sc = (v16h)(_Float16)0.1803368867f;   // 0.125 * log2(e)
#pragma unroll
        for (int i = 0; i < 2; i++) {
            const long qrow = ((long)b * S_ + q0 + i * 16 + l16) * E_ + h * DH_;
            const h8* p0 = (const h8*)(Q + qrow + kAoff);
            aq[i][0].h[0] = p0[0]; aq[i][0].h[1] = p0[1];
            const h8* p1 = (const h8*)(Q + qrow + 32 + kAoff);
            aq[i][1].h[0] = p1[0]; aq[i][1].h[1] = p1[1];
            aq[i][0].v = aq[i][0].v * sc;
            aq[i][1].v = aq[i][1].v * sc;
        }
    }

    float lsum[2][8];                          // per-thread partial row sums
    v8f o_acc[2][4];
#pragma unroll
    for (int i = 0; i < 2; i++) {
#pragma unroll
        for (int r = 0; r < 8; r++) lsum[i][r] = 0.0f;
#pragma unroll
        for (int j = 0; j < 4; j++) o_acc[i][j] = (v8f)0.0f;
    }

    const long vbase = ((long)b * H_ + h) * DH_ * (long)S_;

    for (int t0 = 0; t0 < S_; t0 += 32) {
        // ---- K fragments for this 32-key chunk (shared by both q sub-tiles)
        const long k0 = ((long)b * S_ + t0 + l16) * E_ + h * DH_ + kBoff;
        const long k1 = ((long)b * S_ + t0 + 16 + l16) * E_ + h * DH_ + kBoff;
        v16h bk00 = *(const v16h*)(Km + k0);
        v16h bk01 = *(const v16h*)(Km + k0 + 32);
        v16h bk10 = *(const v16h*)(Km + k1);
        v16h bk11 = *(const v16h*)(Km + k1 + 32);

#pragma unroll
        for (int i = 0; i < 2; i++) {
            // ---- scores (pre-scaled to base-2 via Q) ----
            v8f c0 = (v8f)0.0f, c1 = (v8f)0.0f;
            c0 = wmma_f16(aq[i][0].v, bk00, c0);
            c0 = wmma_f16(aq[i][1].v, bk01, c0);
            c1 = wmma_f16(aq[i][0].v, bk10, c1);
            c1 = wmma_f16(aq[i][1].v, bk11, c1);
            // ---- exp2 + partial row sums + store P to LDS in A staging ----
            _Float16* pl = lds_p[i];
#pragma unroll
            for (int r = 0; r < 8; r++) {
                const float e0 = __builtin_amdgcn_exp2f(c0[r]);
                const float e1 = __builtin_amdgcn_exp2f(c1[r]);
                lsum[i][r] += e0 + e1;
                pl[(mrow + r) * 32 + l16]      = (_Float16)e0;
                pl[(mrow + r) * 32 + 16 + l16] = (_Float16)e1;
            }
        }
        __syncthreads();   // single-wave workgroup: S_NOP + compiler ordering
        H16 ap0, ap1;
        {
            const h8* pp0 = (const h8*)(lds_p[0] + l16 * 32 + kAoff);
            ap0.h[0] = pp0[0]; ap0.h[1] = pp0[1];
            const h8* pp1 = (const h8*)(lds_p[1] + l16 * 32 + kAoff);
            ap1.h[0] = pp1[0]; ap1.h[1] = pp1[1];
        }
        __syncthreads();
        // ---- PV: accumulate 32 x 64 output, V frags shared by both tiles ----
#pragma unroll
        for (int j = 0; j < 4; j++) {
            v16h bv = *(const v16h*)(Vt + vbase + (long)(j * 16 + l16) * S_ + t0 + kBoff);
            o_acc[0][j] = wmma_f16(ap0.v, bv, o_acc[0][j]);
            o_acc[1][j] = wmma_f16(ap1.v, bv, o_acc[1][j]);
        }
    }

    // ---- one-time row-sum reduction across the 16 lanes holding each row ----
#pragma unroll
    for (int i = 0; i < 2; i++) {
#pragma unroll
        for (int r = 0; r < 8; r++) {
            float s = lsum[i][r];
            s += __shfl_xor(s, 1, 32);
            s += __shfl_xor(s, 2, 32);
            s += __shfl_xor(s, 4, 32);
            s += __shfl_xor(s, 8, 32);
            const float inv = 1.0f / s;
#pragma unroll
            for (int j = 0; j < 4; j++) {
                const float v = o_acc[i][j][r] * inv;
                O[((long)b * S_ + q0 + i * 16 + mrow + r) * E_ + h * DH_ + j * 16 + l16] =
                    (_Float16)v;
            }
        }
    }
}

// --------------- residual add + adaptive norm (per-row over E) -------------
__global__ __launch_bounds__(256) void norm_residual(const float* __restrict__ X,
                                                     const float* __restrict__ R,
                                                     const float* __restrict__ aVec,
                                                     const float* __restrict__ bVec,
                                                     float* __restrict__ outF,
                                                     _Float16* __restrict__ outH) {
    const int row = blockIdx.x;            // b*S + s
    const int s = row & (S_ - 1);
    __shared__ float red[2][8];
    const long base = (long)row * E_;
    float vals[4], sum = 0.0f, sq = 0.0f;
#pragma unroll
    for (int i = 0; i < 4; i++) {
        const long idx = base + threadIdx.x + i * 256;
        const float v = X[idx] + R[idx];
        vals[i] = v; sum += v; sq += v * v;
    }
    for (int m = 1; m < 32; m <<= 1) {
        sum += __shfl_xor(sum, m, 32);
        sq  += __shfl_xor(sq,  m, 32);
    }
    const int wave = threadIdx.x >> 5, lane = threadIdx.x & 31;
    if (lane == 0) { red[0][wave] = sum; red[1][wave] = sq; }
    __syncthreads();
    if (wave == 0) {
        float s1 = (lane < 8) ? red[0][lane] : 0.0f;
        float s2 = (lane < 8) ? red[1][lane] : 0.0f;
        for (int m = 1; m < 8; m <<= 1) {
            s1 += __shfl_xor(s1, m, 32);
            s2 += __shfl_xor(s2, m, 32);
        }
        if (lane == 0) { red[0][0] = s1; red[1][0] = s2; }
    }
    __syncthreads();
    const float mean = red[0][0] * (1.0f / E_);
    const float var  = red[1][0] * (1.0f / E_) - mean * mean;
    const float rstd = rsqrtf(var + EPS_);
    const float ga = aVec[s], gb = bVec[s];
#pragma unroll
    for (int i = 0; i < 4; i++) {
        const long idx = base + threadIdx.x + i * 256;
        const float v = (vals[i] - mean) * rstd * ga + gb;
        outF[idx] = v;
        if (outH) outH[idx] = (_Float16)v;
    }
}

// ------------------------------- launcher ----------------------------------
#define MB(x) ((size_t)(x) * 1024 * 1024)

extern "C" void kernel_launch(void* const* d_in, const int* in_sizes, int n_in,
                              void* d_out, int out_size, void* d_ws, size_t ws_size,
                              hipStream_t stream) {
    (void)in_sizes; (void)n_in; (void)out_size; (void)ws_size;
    const float* x    = (const float*)d_in[0];
    const float* Wq   = (const float*)d_in[1];
    const float* bq   = (const float*)d_in[2];
    const float* Wk   = (const float*)d_in[3];
    const float* bk   = (const float*)d_in[4];
    const float* Wv   = (const float*)d_in[5];
    const float* bv   = (const float*)d_in[6];
    const float* Wo   = (const float*)d_in[7];
    const float* bo   = (const float*)d_in[8];
    const float* an1a = (const float*)d_in[9];
    const float* an1b = (const float*)d_in[10];
    const float* W1   = (const float*)d_in[11];
    const float* b1   = (const float*)d_in[12];
    const float* W2   = (const float*)d_in[13];
    const float* b2   = (const float*)d_in[14];
    const float* an2a = (const float*)d_in[15];
    const float* an2b = (const float*)d_in[16];

    char* ws = (char*)d_ws;
    _Float16* Xh   = (_Float16*)(ws + MB(0));    // 16 MB
    _Float16* Wqh  = (_Float16*)(ws + MB(16));   // 2 MB
    _Float16* Wkh  = (_Float16*)(ws + MB(18));
    _Float16* Wvh  = (_Float16*)(ws + MB(20));
    _Float16* Woh  = (_Float16*)(ws + MB(22));
    _Float16* W1h  = (_Float16*)(ws + MB(24));   // 4 MB
    _Float16* W2h  = (_Float16*)(ws + MB(28));   // 4 MB
    _Float16* Qh   = (_Float16*)(ws + MB(32));   // 16 MB
    _Float16* Kh   = (_Float16*)(ws + MB(48));   // 16 MB
    _Float16* Vt   = (_Float16*)(ws + MB(64));   // 16 MB (B,H,Dh,S)
    _Float16* Ah   = (_Float16*)(ws + MB(80));   // 16 MB attention out f16
    float*    F32s = (float*)   (ws + MB(96));   // 32 MB (Wo out, later FFN out)
    float*    Hf   = (float*)   (ws + MB(128));  // 32 MB h (post norm1, f32)
    _Float16* Hh   = (_Float16*)(ws + MB(160));  // 16 MB h (f16)
    _Float16* FF1h = (_Float16*)(ws + MB(176));  // 32 MB

    const long nX = (long)M_ * E_;
    const long nW = (long)E_ * E_;
    const long nW12 = (long)2 * E_ * E_;

    // f32 -> f16 conversions
    cvt_f32_f16<<<(nX + 255) / 256, 256, 0, stream>>>(x,  Xh,  nX);
    cvt_f32_f16<<<(nW + 255) / 256, 256, 0, stream>>>(Wq, Wqh, nW);
    cvt_f32_f16<<<(nW + 255) / 256, 256, 0, stream>>>(Wk, Wkh, nW);
    cvt_f32_f16<<<(nW + 255) / 256, 256, 0, stream>>>(Wv, Wvh, nW);
    cvt_f32_f16<<<(nW + 255) / 256, 256, 0, stream>>>(Wo, Woh, nW);
    cvt_f32_f16<<<(nW12 + 255) / 256, 256, 0, stream>>>(W1, W1h, nW12);
    cvt_f32_f16<<<(nW12 + 255) / 256, 256, 0, stream>>>(W2, W2h, nW12);

    // QKV projections
    dim3 blk(256);
    dim3 gE(E_ / 256, M_ / 128);         // (4, 64)
    gemm_wmma<1, 0, 0><<<gE, blk, 0, stream>>>(Xh, Wqh, bq, Qh, M_, E_, E_);
    gemm_wmma<1, 0, 0><<<gE, blk, 0, stream>>>(Xh, Wkh, bk, Kh, M_, E_, E_);
    gemm_wmma<1, 0, 1><<<gE, blk, 0, stream>>>(Xh, Wvh, bv, Vt, M_, E_, E_);

    // attention: one wave per (b, h, 32-row query tile)
    attn_wmma<<<B_ * H_ * (S_ / 32), 32, 0, stream>>>(Qh, Kh, Vt, Ah);

    // output projection (f32) -> residual + adaptive norm 1
    gemm_wmma<0, 0, 0><<<gE, blk, 0, stream>>>(Ah, Woh, bo, F32s, M_, E_, E_);
    norm_residual<<<M_, 256, 0, stream>>>(F32s, x, an1a, an1b, Hf, Hh);

    // FFN: relu(h @ W1^T + b1) @ W2^T + b2 -> residual + adaptive norm 2
    dim3 g2E(2 * E_ / 256, M_ / 128);    // (8, 64)
    gemm_wmma<1, 1, 0><<<g2E, blk, 0, stream>>>(Hh, W1h, b1, FF1h, M_, 2 * E_, E_);
    gemm_wmma<0, 0, 0><<<gE, blk, 0, stream>>>(FF1h, W2h, b2, F32s, M_, E_, 2 * E_);
    norm_residual<<<M_, 256, 0, stream>>>(F32s, Hf, an2a, an2b, (float*)d_out, nullptr);
}
